// GPSA2D_3590592659850
// MI455X (gfx1250) — compile-verified, matrix operand order checked by reading, not run on hardware
//
#include <hip/hip_runtime.h>
#include <stdint.h>

typedef __attribute__((ext_vector_type(16))) __bf16 v16bf;
typedef __attribute__((ext_vector_type(8)))  float  v8f;
typedef unsigned short u16;
typedef unsigned int   u32;

union Frag16 {
  v16bf v;
  uint4 q[2];
};

__device__ __forceinline__ u16 f2bf(float f) {
  u32 u = __float_as_uint(f);
  u32 r = (u + 0x7FFFu + ((u >> 16) & 1u)) >> 16;   // round-to-nearest-even
  return (u16)r;
}

// Load one 16x32 bf16 operand tile (A: lane=row M, or B: lane=col N; elements
// run over K). ISA 16-bit layout: lane half lh picks K in {8*lh..8*lh+7} (v0-3)
// and {16+8*lh..23+8*lh} (v4-7) -> two 16B vector loads per lane.
__device__ __forceinline__ v16bf load_frag(const u16* base, int stride, int lane) {
  int m = lane & 15, lh = lane >> 4;
  Frag16 f;
  f.q[0] = *(const uint4*)(base + (size_t)m * stride + 8 * lh);
  f.q[1] = *(const uint4*)(base + (size_t)m * stride + 16 + 8 * lh);
  return f.v;
}

// One 16B async DMA: global -> LDS, tracked by ASYNCcnt (no VGPRs held in flight).
// Generic LDS pointer low 32 bits == LDS byte address (ISA 10.2 aperture map).
__device__ __forceinline__ void async_cp16(u16* lds, const u16* g) {
  unsigned lo = (unsigned)(uintptr_t)lds;
  unsigned long long ga = (unsigned long long)(uintptr_t)g;
  asm volatile("global_load_async_to_lds_b128 %0, %1, off" :: "v"(lo), "v"(ga) : "memory");
}
__device__ __forceinline__ void wait_async0() {
  asm volatile("s_wait_asynccnt 0" ::: "memory");
}

__global__ void k_zero16(u16* p, int n) {
  int i = blockIdx.x * blockDim.x + threadIdx.x;
  if (i < n) p[i] = 0;
}

__global__ void k_cvt_bf16(const float* __restrict__ in, u16* __restrict__ out, int n) {
  int i = blockIdx.x * blockDim.x + threadIdx.x;
  if (i < n) out[i] = f2bf(in[i]);
}

// ---------------- QKV projection: (3200x384) @ W^T, W row-major [c][k] ------
// Writes Q,K as [b][h][tok][32] (d padded 24->32), V transposed [b][h][32][tok].
__global__ void k_qkv(const u16* __restrict__ xb,
                      const u16* __restrict__ Wq, const u16* __restrict__ Wk,
                      const u16* __restrict__ Wv,
                      u16* __restrict__ qw, u16* __restrict__ kw, u16* __restrict__ vT) {
  int rt = blockIdx.x, ct = blockIdx.y, which = blockIdx.z;
  int lane = threadIdx.x;
  const u16* W = (which == 0) ? Wq : ((which == 1) ? Wk : Wv);
  const u16* abase = xb + (size_t)rt * 16 * 384;
  const u16* bbase = W  + (size_t)ct * 16 * 384;
  v8f acc = {};
  for (int kt = 0; kt < 12; ++kt) {
    if (kt < 10) {  // stream hint two K-tiles ahead -> global_prefetch
      __builtin_prefetch(abase + (kt + 2) * 32, 0, 0);
      __builtin_prefetch(bbase + (kt + 2) * 32, 0, 0);
    }
    v16bf a = load_frag(abase + kt * 32, 384, lane);
    v16bf b = load_frag(bbase + kt * 32, 384, lane);
    acc = __builtin_amdgcn_wmma_f32_16x16x32_bf16(false, a, false, b, (short)0, acc, false, false);
  }
  int n = lane & 15, lh = lane >> 4;
  int c = ct * 16 + n, head = c / 24, dcol = c % 24;
  for (int vr = 0; vr < 8; ++vr) {
    int r = rt * 16 + vr + 8 * lh;          // global row in [0,3200)
    int bidx = (r >= 1600) ? 1 : 0;
    int tok = r - bidx * 1600;
    int bh = bidx * 16 + head;
    u16 val = f2bf(acc[vr]);
    if (which == 2) {
      vT[((size_t)bh * 32 + dcol) * 1600 + tok] = val;
    } else {
      u16* dst = (which == 0) ? qw : kw;
      dst[((size_t)bh * 1600 + tok) * 32 + dcol] = val;
    }
  }
}

// ---------------- fused attention per (b, head, 16-row tile) ----------------
// LDS: [0,102400)   KV   staging window (K strip in pass 1, V strip in pass 3)
//      [102400,..)  S    16x1600 f32 content scores
//      [204800,..)  At   16x1600 bf16 blended probabilities
//      [256000,..)  Oc   16x32 f32 cross-wave output accumulator
__global__ void __launch_bounds__(256) k_attn(
    const u16* __restrict__ qw, const u16* __restrict__ kw, const u16* __restrict__ vT,
    const float* __restrict__ Wpos, const float* __restrict__ bpos,
    const float* __restrict__ gating, u16* __restrict__ ctx) {
  extern __shared__ char smem[];
  u16*   KV = (u16*)smem;
  float* S  = (float*)(smem + 102400);
  u16*   At = (u16*)(smem + 204800);
  float* Oc = (float*)(smem + 256000);

  const int rt = blockIdx.x, head = blockIdx.y, b = blockIdx.z;
  const int bh = b * 16 + head;
  const int lane = threadIdx.x & 31, wave = threadIdx.x >> 5;
  const int n = lane & 15, lh = lane >> 4;
  const float SCALE = 0.20412414523193154f;           // 24^-0.5

  // ---- async DMA: whole K strip [1600][32] (102,400 B) into LDS ----
  {
    const u16* kg = kw + (size_t)bh * 1600 * 32;
    for (int it = 0; it < 25; ++it) {
      int e = (it * 256 + (int)threadIdx.x) * 8;      // 16B per lane per issue
      async_cp16(KV + e, kg + e);
    }
  }
  wait_async0();
  __syncthreads();

  // ---- pass 1: content scores Q Kt into LDS (100 WMMAs / block) ----
  v16bf qf = load_frag(qw + ((size_t)bh * 1600 + rt * 16) * 32, 32, lane);
  for (int jt = wave; jt < 100; jt += 8) {
    v16bf kf = load_frag(KV + jt * 16 * 32, 32, lane);   // ds_load_b128
    v8f c = {};
    c = __builtin_amdgcn_wmma_f32_16x16x32_bf16(false, qf, false, kf, (short)0, c, false, false);
    for (int vr = 0; vr < 8; ++vr)
      S[(vr + 8 * lh) * 1600 + jt * 16 + n] = c[vr] * SCALE;
  }
  __syncthreads();

  // ---- kick off V strip DMA now; it overlaps the whole softmax pass ----
  {
    const u16* vg = vT + (size_t)bh * 32 * 1600;     // [dcol(32)][tok(1600)]
    for (int it = 0; it < 25; ++it) {
      int e = (it * 256 + (int)threadIdx.x) * 8;
      async_cp16(KV + e, vg + e);
    }
  }

  // ---- pass 2: dual softmax + gated blend, positional logits on the fly ----
  const float w0 = Wpos[head * 3 + 0], w1 = Wpos[head * 3 + 1], w2 = Wpos[head * 3 + 2];
  const float bp = bpos[head];
  const float g = 1.f / (1.f + __expf(-gating[head]));
  for (int row = wave; row < 16; row += 8) {
    int i = rt * 16 + row;
    int iy = i / 40, ix = i % 40;
    float mS = -1e30f, mP = -1e30f;
    for (int j = lane; j < 1600; j += 32) {
      float s = S[row * 1600 + j];
      float dx = (float)(j % 40 - ix), dy = (float)(j / 40 - iy);
      float p = w0 * dx + w1 * dy + w2 * (dx * dx + dy * dy) + bp;
      mS = fmaxf(mS, s); mP = fmaxf(mP, p);
    }
    for (int o = 16; o; o >>= 1) {
      mS = fmaxf(mS, __shfl_xor(mS, o, 32));
      mP = fmaxf(mP, __shfl_xor(mP, o, 32));
    }
    float sS = 0.f, sP = 0.f;
    for (int j = lane; j < 1600; j += 32) {
      float s = S[row * 1600 + j];
      float dx = (float)(j % 40 - ix), dy = (float)(j / 40 - iy);
      float p = w0 * dx + w1 * dy + w2 * (dx * dx + dy * dy) + bp;
      sS += __expf(s - mS); sP += __expf(p - mP);
    }
    for (int o = 16; o; o >>= 1) {
      sS += __shfl_xor(sS, o, 32);
      sP += __shfl_xor(sP, o, 32);
    }
    // blended attn sums to (1-g)+g == 1, so the reference renormalize is identity
    float aS = (1.f - g) / sS, aP = g / sP;
    for (int j = lane; j < 1600; j += 32) {
      float s = S[row * 1600 + j];
      float dx = (float)(j % 40 - ix), dy = (float)(j / 40 - iy);
      float p = w0 * dx + w1 * dy + w2 * (dx * dx + dy * dy) + bp;
      At[row * 1600 + j] = f2bf(__expf(s - mS) * aS + __expf(p - mP) * aP);
    }
  }
  Oc[threadIdx.x] = 0.f;
  Oc[threadIdx.x + 256] = 0.f;
  wait_async0();           // V strip resident in LDS
  __syncthreads();

  // ---- pass 3: attn @ V (100 WMMAs / block), all operands from LDS ----
  v8f a0 = {}, a1 = {};
  for (int jc = wave; jc < 50; jc += 8) {
    int jb = jc * 32;
    v16bf af = load_frag(At + jb, 1600, lane);
    v16bf b0 = load_frag(KV + jb, 1600, lane);               // dcol 0..15
    v16bf b1 = load_frag(KV + 16 * 1600 + jb, 1600, lane);   // dcol 16..31
    a0 = __builtin_amdgcn_wmma_f32_16x16x32_bf16(false, af, false, b0, (short)0, a0, false, false);
    a1 = __builtin_amdgcn_wmma_f32_16x16x32_bf16(false, af, false, b1, (short)0, a1, false, false);
  }
  for (int vr = 0; vr < 8; ++vr) {
    int m = vr + 8 * lh;
    atomicAdd(&Oc[m * 32 + n], a0[vr]);
    atomicAdd(&Oc[m * 32 + 16 + n], a1[vr]);
  }
  __syncthreads();

  for (int t = threadIdx.x; t < 384; t += 256) {
    int m = t / 24, dc = t % 24;
    ctx[((size_t)b * 1600 + rt * 16 + m) * 384 + head * 24 + dc] = f2bf(Oc[m * 32 + dc]);
  }
}

// ---------------- output projection: ctx @ Wproj^T + bproj (fp32 out) -------
__global__ void k_proj(const u16* __restrict__ ctxb, const u16* __restrict__ Wp,
                       const float* __restrict__ bproj, float* __restrict__ out) {
  int rt = blockIdx.x, ct = blockIdx.y, lane = threadIdx.x;
  const u16* abase = ctxb + (size_t)rt * 16 * 384;
  const u16* bbase = Wp   + (size_t)ct * 16 * 384;
  v8f acc = {};
  for (int kt = 0; kt < 12; ++kt) {
    if (kt < 10) {
      __builtin_prefetch(abase + (kt + 2) * 32, 0, 0);
      __builtin_prefetch(bbase + (kt + 2) * 32, 0, 0);
    }
    v16bf a = load_frag(abase + kt * 32, 384, lane);
    v16bf b = load_frag(bbase + kt * 32, 384, lane);
    acc = __builtin_amdgcn_wmma_f32_16x16x32_bf16(false, a, false, b, (short)0, acc, false, false);
  }
  int n = lane & 15, lh = lane >> 4;
  float bias = bproj[ct * 16 + n];
  for (int vr = 0; vr < 8; ++vr) {
    int r = rt * 16 + vr + 8 * lh;
    out[(size_t)r * 384 + ct * 16 + n] = acc[vr] + bias;
  }
}

extern "C" void kernel_launch(void* const* d_in, const int* in_sizes, int n_in,
                              void* d_out, int out_size, void* d_ws, size_t ws_size,
                              hipStream_t stream) {
  (void)in_sizes; (void)n_in; (void)out_size; (void)ws_size;
  const float* x      = (const float*)d_in[0];
  const float* Wq     = (const float*)d_in[1];
  const float* Wk     = (const float*)d_in[2];
  const float* Wv     = (const float*)d_in[3];
  const float* Wpos   = (const float*)d_in[4];
  const float* bpos   = (const float*)d_in[5];
  const float* gating = (const float*)d_in[6];
  const float* Wproj  = (const float*)d_in[7];
  const float* bproj  = (const float*)d_in[8];
  float* out = (float*)d_out;

  // workspace layout (bytes), all 16B-aligned
  char* ws = (char*)d_ws;
  u16* xb  = (u16*)(ws + 0);          // 3200 x 384 bf16        (2,457,600)
  u16* Wqb = (u16*)(ws + 2457600);    // 384 x 384 bf16
  u16* Wkb = (u16*)(ws + 2752512);
  u16* Wvb = (u16*)(ws + 3047424);
  u16* Wpb = (u16*)(ws + 3342336);
  u16* qw  = (u16*)(ws + 3637248);    // [2][16][1600][32] bf16 (3,276,800)
  u16* kw  = (u16*)(ws + 6914048);    // [2][16][1600][32] bf16
  u16* vT  = (u16*)(ws + 10190848);   // [2][16][32][1600] bf16
  u16* ctx = (u16*)(ws + 13467648);   // [2][1600][384] bf16    (2,457,600)

  k_cvt_bf16<<<(1228800 + 255) / 256, 256, 0, stream>>>(x, xb, 1228800);
  k_cvt_bf16<<<(147456 + 255) / 256, 256, 0, stream>>>(Wq, Wqb, 147456);
  k_cvt_bf16<<<(147456 + 255) / 256, 256, 0, stream>>>(Wk, Wkb, 147456);
  k_cvt_bf16<<<(147456 + 255) / 256, 256, 0, stream>>>(Wv, Wvb, 147456);
  k_cvt_bf16<<<(147456 + 255) / 256, 256, 0, stream>>>(Wproj, Wpb, 147456);

  // q/k/vT are contiguous: one zero pass covers the d-padding lanes
  k_zero16<<<(4915200 + 255) / 256, 256, 0, stream>>>(qw, 4915200);

  k_qkv<<<dim3(200, 24, 3), 32, 0, stream>>>(xb, Wqb, Wkb, Wvb, qw, kw, vT);

  // dynamic LDS: 102400 (KV) + 102400 (S) + 51200 (At) + 2048 (Oc) = 258,048 B
  k_attn<<<dim3(100, 16, 2), 256, 258048, stream>>>(qw, kw, vT, Wpos, bpos, gating, ctx);

  k_proj<<<dim3(200, 24), 32, 0, stream>>>(ctx, Wpb, bproj, out);
}
